// ArcFaceHead_78142634983760
// MI455X (gfx1250) — matrix operand order repaired and checked by reading, not compiled, not run on hardware
//
#include <hip/hip_runtime.h>
#include <math.h>

typedef __attribute__((ext_vector_type(16))) _Float16 v16h;
typedef __attribute__((ext_vector_type(8)))  float    v8f;
typedef __attribute__((ext_vector_type(4)))  _Float16 v4h;
typedef __attribute__((ext_vector_type(4)))  int      i32x4;

#define S_SCALE 30.0f
#define MARGIN  0.3f
#define EPS_CLIP 1e-7f

#define DDIM 512      // K dimension
#define BM   128      // block tile M
#define BN   128      // block tile N
#define BK   32       // K step (matches 16x16x32 WMMA)
#define LDT  40       // padded LDS stride in f16 elements (40*2=80B, 16B aligned)
#define KSTEPS (DDIM / BK)

#if __has_builtin(__builtin_amdgcn_global_load_async_to_lds_b128)
#define HAVE_ASYNC_LDS 1
#else
#define HAVE_ASYNC_LDS 0
#endif

__device__ __forceinline__ void wait_asynccnt0() {
#if __has_builtin(__builtin_amdgcn_s_wait_asynccnt)
    __builtin_amdgcn_s_wait_asynccnt(0);
#else
    asm volatile("s_wait_asynccnt 0x0" ::: "memory");
#endif
}

// ---------------------------------------------------------------------------
// Pass 1: L2-normalize rows of a [srcRows, 512] f32 matrix into an f16 copy
// of [rows, 512]; rows beyond srcRows are zero-filled (N-tile padding).
// One wave (32 lanes) per row; 8 rows per 256-thread block.
// ---------------------------------------------------------------------------
__global__ __launch_bounds__(256) void normalize_rows_f16(
    const float* __restrict__ src, _Float16* __restrict__ dst,
    int rows, int srcRows)
{
    int row  = blockIdx.x * 8 + (threadIdx.x >> 5);
    int lane = threadIdx.x & 31;
    if (row >= rows) return;

    v4h* d = (v4h*)(dst + (size_t)row * DDIM);

    if (row >= srcRows) {           // zero-fill padding rows
        v4h z = {};
#pragma unroll
        for (int i = 0; i < 4; ++i) d[lane + i * 32] = z;
        return;
    }

    const float4* s = (const float4*)(src + (size_t)row * DDIM);
    float4 v[4];
    float ss = 0.0f;
#pragma unroll
    for (int i = 0; i < 4; ++i) {
        v[i] = s[lane + i * 32];
        ss += v[i].x * v[i].x + v[i].y * v[i].y + v[i].z * v[i].z + v[i].w * v[i].w;
    }
#pragma unroll
    for (int off = 16; off > 0; off >>= 1)
        ss += __shfl_xor(ss, off, 32);

    float inv = 1.0f / fmaxf(sqrtf(ss), 1e-12f);

#pragma unroll
    for (int i = 0; i < 4; ++i) {
        v4h h;
        h.x = (_Float16)(v[i].x * inv);
        h.y = (_Float16)(v[i].y * inv);
        h.z = (_Float16)(v[i].z * inv);
        h.w = (_Float16)(v[i].w * inv);
        d[lane + i * 32] = h;
    }
}

// ---------------------------------------------------------------------------
// Pass 2: tiled f16 WMMA GEMM: out[m,n] = S * clip(Fn[m,:] . Wn[n,:])
// 8 waves per block, each wave owns a 64x32 sub-tile = 4x2 WMMA fragments.
// Double-buffered LDS tiles filled with async global->LDS copies.
// W16 is padded to a multiple of BN rows -> no bounds checks in the hot loop.
// ---------------------------------------------------------------------------
__global__ __launch_bounds__(256) void arcface_gemm(
    const _Float16* __restrict__ A16,  // [Brows, 512] normalized features
    const _Float16* __restrict__ W16,  // [Cpad, 512] normalized weights (padded)
    float* __restrict__ out, int Brows, int C)
{
    __shared__ _Float16 sA[2][BM * LDT];
    __shared__ _Float16 sB[2][BN * LDT];

    const int tid  = threadIdx.x;
    const int lane = tid & 31;
    const int wid  = tid >> 5;
    const int wm   = (wid >> 2) * 64;   // wave M offset in tile: 0 or 64
    const int wn   = (wid & 3) * 32;    // wave N offset in tile: 0..96
    const int l16  = lane & 15;
    const int lhi  = lane >> 4;         // 0 for lanes 0-15, 1 for 16-31

    const int blockM = blockIdx.y * BM;
    const int blockN = blockIdx.x * BN;

    // Per-thread copy map: 512 chunks of 16B per tile, 2 chunks per thread.
    const _Float16* gA[2];
    const _Float16* gB[2];
    int ldsOff[2];
#pragma unroll
    for (int i = 0; i < 2; ++i) {
        int chunk = tid + i * 256;
        int row   = chunk >> 2;          // 0..127
        int off   = (chunk & 3) * 8;     // f16 offset 0,8,16,24
        int am    = blockM + row;
        am        = am < Brows ? am : (Brows - 1);   // clamp, branch-free
        gA[i] = A16 + (size_t)am * DDIM + off;
        gB[i] = W16 + (size_t)(blockN + row) * DDIM + off;  // Cpad-padded: in-bounds
        ldsOff[i] = row * LDT + off;
    }

    v8f acc[4][2];
#pragma unroll
    for (int mi = 0; mi < 4; ++mi)
#pragma unroll
        for (int nj = 0; nj < 2; ++nj)
#pragma unroll
            for (int e = 0; e < 8; ++e)
                acc[mi][nj][e] = 0.0f;

    union FH { uint4 u[2]; v16h h; };

#if HAVE_ASYNC_LDS
    typedef __attribute__((address_space(1))) i32x4 gi32x4_t;  // global int4
    typedef __attribute__((address_space(3))) i32x4 li32x4_t;  // shared int4

    // Issue async copies of tile `t` (K offset t*BK) into buffer `buf`.
    auto issue_tile = [&](int buf, int k0) {
#pragma unroll
        for (int i = 0; i < 2; ++i) {
            __builtin_amdgcn_global_load_async_to_lds_b128(
                (gi32x4_t*)(gA[i] + k0), (li32x4_t*)(&sA[buf][ldsOff[i]]), 0, 0);
            __builtin_amdgcn_global_load_async_to_lds_b128(
                (gi32x4_t*)(gB[i] + k0), (li32x4_t*)(&sB[buf][ldsOff[i]]), 0, 0);
        }
    };

    issue_tile(0, 0);   // prologue: tile 0 in flight

    for (int t = 0; t < KSTEPS; ++t) {
        const int buf = t & 1;
        wait_asynccnt0();       // this thread's tile-t copies landed in LDS
        __syncthreads();        // everyone's copies landed; prev-buf reads done

        const _Float16* a = sA[buf];
        const _Float16* b = sB[buf];

        // B fragments: 16 contiguous K per column; kbase = lhi*16
        FH fb[2];
#pragma unroll
        for (int nj = 0; nj < 2; ++nj) {
            const uint4* p = (const uint4*)(b + (wn + nj * 16 + l16) * LDT + lhi * 16);
            fb[nj].u[0] = p[0];
            fb[nj].u[1] = p[1];
        }
        // A fragments + WMMA; 16 contiguous K per row; kbase = lhi*16
#pragma unroll
        for (int mi = 0; mi < 4; ++mi) {
            FH fa;
            const _Float16* ap = a + (wm + mi * 16 + l16) * LDT + lhi * 16;
            fa.u[0] = *(const uint4*)(ap);
            fa.u[1] = *(const uint4*)(ap + 8);
#pragma unroll
            for (int nj = 0; nj < 2; ++nj) {
                acc[mi][nj] = __builtin_amdgcn_wmma_f32_16x16x32_f16(
                    false, fa.h, false, fb[nj].h,
                    (short)0, acc[mi][nj], false, false);
            }
        }

        // Prefetch tile t+1 into the other buffer. Issued after this tile's
        // fragment reads, and the other buffer's readers all consumed their
        // data before the barrier above -> race-free single-barrier pipeline.
        if (t + 1 < KSTEPS)
            issue_tile(buf ^ 1, (t + 1) * BK);
    }
#else
    // Fallback: synchronous global->reg->LDS staging, single buffer.
    for (int k0 = 0; k0 < DDIM; k0 += BK) {
        uint4 ga[2], gb[2];
#pragma unroll
        for (int i = 0; i < 2; ++i) {
            ga[i] = *(const uint4*)(gA[i] + k0);
            gb[i] = *(const uint4*)(gB[i] + k0);
        }
        __syncthreads();
#pragma unroll
        for (int i = 0; i < 2; ++i) {
            *(uint4*)(&sA[0][ldsOff[i]]) = ga[i];
            *(uint4*)(&sB[0][ldsOff[i]]) = gb[i];
        }
        __syncthreads();

        FH fb[2];
#pragma unroll
        for (int nj = 0; nj < 2; ++nj) {
            const uint4* p = (const uint4*)(&sB[0][(wn + nj * 16 + l16) * LDT + lhi * 16]);
            fb[nj].u[0] = p[0];
            fb[nj].u[1] = p[1];
        }
#pragma unroll
        for (int mi = 0; mi < 4; ++mi) {
            FH fa;
            const _Float16* ap = &sA[0][(wm + mi * 16 + l16) * LDT + lhi * 16];
            fa.u[0] = *(const uint4*)(ap);
            fa.u[1] = *(const uint4*)(ap + 8);
#pragma unroll
            for (int nj = 0; nj < 2; ++nj) {
                acc[mi][nj] = __builtin_amdgcn_wmma_f32_16x16x32_f16(
                    false, fa.h, false, fb[nj].h,
                    (short)0, acc[mi][nj], false, false);
            }
        }
    }
#endif

    // ---- epilogue: clip to (-1+eps, 1-eps), scale by S, store ----
    const float lo = -1.0f + EPS_CLIP;
    const float hi =  1.0f - EPS_CLIP;
#pragma unroll
    for (int mi = 0; mi < 4; ++mi) {
#pragma unroll
        for (int nj = 0; nj < 2; ++nj) {
            int col = blockN + wn + nj * 16 + l16;
            if (col < C) {
                int m0 = blockM + wm + mi * 16 + lhi * 8;
#pragma unroll
                for (int e = 0; e < 8; ++e) {
                    int m = m0 + e;
                    if (m < Brows) {
                        float v = acc[mi][nj][e];
                        v = fminf(fmaxf(v, lo), hi);
                        out[(size_t)m * C + col] = S_SCALE * v;
                    }
                }
            }
        }
    }
}

// ---------------------------------------------------------------------------
// Pass 3: apply angular margin to the target column of each row.
// cos(acos(c)+M) = c*cos(M) - sqrt(1-c^2)*sin(M)
// ---------------------------------------------------------------------------
__global__ __launch_bounds__(256) void patch_margin(
    const int* __restrict__ labels, float* __restrict__ out, int Brows, int C)
{
    int b = blockIdx.x * 256 + threadIdx.x;
    if (b >= Brows) return;
    int c = labels[b];
    if (c < 0 || c >= C) return;
    size_t idx = (size_t)b * C + c;
    float v  = out[idx] * (1.0f / S_SCALE);   // already clipped
    float sm = sqrtf(fmaxf(1.0f - v * v, 0.0f));
    float cm = v * cosf(MARGIN) - sm * sinf(MARGIN);
    out[idx] = S_SCALE * cm;
}

// ---------------------------------------------------------------------------
extern "C" void kernel_launch(void* const* d_in, const int* in_sizes, int n_in,
                              void* d_out, int out_size, void* d_ws, size_t ws_size,
                              hipStream_t stream)
{
    const float* feats  = (const float*)d_in[0];
    const int*   labels = (const int*)d_in[1];
    const float* weight = (const float*)d_in[2];
    float*       out    = (float*)d_out;

    const int Brows = in_sizes[0] / DDIM;   // 4096
    const int C     = in_sizes[2] / DDIM;   // 50000
    const int Cpad  = (C + BN - 1) / BN * BN;   // 50048

    _Float16* f16 = (_Float16*)d_ws;                   // [Brows, 512]
    _Float16* w16 = f16 + (size_t)Brows * DDIM;        // [Cpad, 512]

    normalize_rows_f16<<<(Brows + 7) / 8, 256, 0, stream>>>(feats, f16, Brows, Brows);
    normalize_rows_f16<<<(Cpad + 7) / 8, 256, 0, stream>>>(weight, w16, Cpad, C);

    dim3 grid(Cpad / BN, (Brows + BM - 1) / BM);
    arcface_gemm<<<grid, 256, 0, stream>>>(f16, w16, out, Brows, C);

    patch_margin<<<(Brows + 255) / 256, 256, 0, stream>>>(labels, out, Brows, C);
}
